// HierarchicalReasoningEncoder_27496380629693
// MI455X (gfx1250) — matrix operand order repaired and checked by reading, not compiled, not run on hardware
//
#include <hip/hip_runtime.h>
#include <hip/hip_bf16.h>

typedef __bf16 bf16_t;
typedef __attribute__((ext_vector_type(16))) __bf16 v16bf;
typedef __attribute__((ext_vector_type(8)))  float  v8f;
typedef __attribute__((ext_vector_type(4)))  unsigned int v4u;
typedef __attribute__((ext_vector_type(8)))  int v8i_t;
typedef __attribute__((ext_vector_type(4)))  int v4i_t;

// ---------- bf16 helpers (RNE convert via bit ops) ----------
__device__ __forceinline__ bf16_t f2bf(float f) {
  unsigned u = __float_as_uint(f);
  unsigned r = u + 0x7FFFu + ((u >> 16) & 1u);
  unsigned short s = (unsigned short)(r >> 16);
  return __builtin_bit_cast(bf16_t, s);
}
__device__ __forceinline__ float bf2f(bf16_t b) {
  unsigned short s = __builtin_bit_cast(unsigned short, b);
  return __uint_as_float(((unsigned)s) << 16);
}

union FragU { v16bf v; uint4 q[2]; };

// A operand (16x32 MxK, bf16), row-major A[m][k]:
// lane%16 = m ; per lane two contiguous 16B chunks at k0 + 8*(lane>>4) and +16.
__device__ __forceinline__ v16bf load_frag_a(const bf16_t* A, int lda, int m0,
                                             int k0, int lane) {
  const bf16_t* p = A + (size_t)(m0 + (lane & 15)) * lda + k0 + ((lane >> 4) << 3);
  FragU u;
  u.q[0] = *(const uint4*)p;
  u.q[1] = *(const uint4*)(p + 16);
  return u.v;
}
// B operand (32x16 KxN) from row-major W[n][k]:
// lane%16 = n ; per lane two contiguous 16B chunks of row n at k0 + 8*(lane>>4) and +16.
__device__ __forceinline__ v16bf load_frag_b(const bf16_t* W, int ldw, int n0,
                                             int k0, int lane) {
  const bf16_t* p = W + (size_t)(n0 + (lane & 15)) * ldw + k0 + ((lane >> 4) << 3);
  FragU u;
  u.q[0] = *(const uint4*)p;
  u.q[1] = *(const uint4*)(p + 16);
  return u.v;
}

__device__ __forceinline__ v8f wmma_bf16(v16bf a, v16bf b, v8f c) {
  return __builtin_amdgcn_wmma_f32_16x16x32_bf16(false, a, false, b,
                                                 (short)0, c, false, false);
}
__device__ __forceinline__ v8f vzero8() {
  v8f z;
#pragma unroll
  for (int i = 0; i < 8; ++i) z[i] = 0.f;
  return z;
}
__device__ __forceinline__ float sigm(float x) { return 1.f / (1.f + __expf(-x)); }

// ---------- Tensor Data Mover: 2D tile (ty rows x tx elems, bf16) -> LDS ----------
// Builds D# per CDNA5 ISA ch8: group0 = {count, lds_addr, global_addr, type=2},
// group1 = {data_size=2B, tensor dims = tile dims, dim0 stride}.
__device__ __forceinline__ unsigned lds_off_of(const void* p) {
  return (unsigned)(unsigned long long)p;  // generic LDS ptr: low 32 bits = LDS offset
}
__device__ __forceinline__ void tdm_load_2d(const void* gptr, unsigned ldsoff,
                                            unsigned tx, unsigned ty,
                                            unsigned long long stride_elems) {
  unsigned long long ga = (unsigned long long)gptr;
  v4u g0;
  g0[0] = 1u;                                        // count=1, no gather
  g0[1] = ldsoff;                                    // lds byte address
  g0[2] = (unsigned)(ga & 0xFFFFFFFFu);              // global_addr[31:0]
  g0[3] = (unsigned)((ga >> 32) & 0x1FFFFFFu) | (2u << 30);  // [56:32] | type=2
  v8i_t g1;
  g1[0] = (int)(1u << 16);                           // data_size=1 -> 2 bytes
  g1[1] = (int)((tx & 0xFFFFu) << 16);               // tensor_dim0[15:0] @bit48
  g1[2] = (int)(((tx >> 16) & 0xFFFFu) | ((ty & 0xFFFFu) << 16));  // d0 hi | d1 lo
  g1[3] = (int)(((ty >> 16) & 0xFFFFu) | (tx << 16));              // d1 hi | tile_dim0
  g1[4] = (int)(ty & 0xFFFFu);                       // tile_dim1, tile_dim2=0
  g1[5] = (int)(stride_elems & 0xFFFFFFFFu);         // dim0_stride[31:0]
  g1[6] = (int)((stride_elems >> 32) & 0xFFFFu);     // dim0_stride[47:32], s1=0
  g1[7] = 0;
  v4i_t z4 = {0, 0, 0, 0};
#if defined(__clang_major__) && (__clang_major__ >= 23)
  v8i_t z8 = {0, 0, 0, 0, 0, 0, 0, 0};
  __builtin_amdgcn_tensor_load_to_lds(g0, g1, z4, z4, z8, 0);
#else
  __builtin_amdgcn_tensor_load_to_lds(g0, g1, z4, z4, 0);
#endif
}

// ---------- pack / convert ----------
__global__ void k_pack_x(const float* __restrict__ tf, bf16_t* __restrict__ X) {
  size_t idx = (size_t)blockIdx.x * blockDim.x + threadIdx.x;  // < 32768*512
  int h = (int)(idx & 511);
  int row = (int)(idx >> 9);
  int t = row & 127;
  int seg = row >> 7;
  int i = seg & 15;
  int b = seg >> 4;
  float v = tf[(((size_t)(i * 128 + t) * 16 + b) << 9) + h];
  X[idx] = f2bf(v);
}

__global__ void k_cvt(const float* __restrict__ src, bf16_t* __restrict__ dst, int n) {
  int i = blockIdx.x * blockDim.x + threadIdx.x;
  if (i < n) dst[i] = f2bf(src[i]);
}
__global__ void k_zero(float* __restrict__ p, int n) {
  int i = blockIdx.x * blockDim.x + threadIdx.x;
  if (i < n) p[i] = 0.f;
}

// ---------- GEMM: out[m][n] = sum_k A[m][k]*W[n][k] + bias(n), bf16 out ----------
// block = 256 (8 waves); tile 128(M) x 128(N). B tile staged in LDS via TDM,
// double buffered; one DMA per K-step shared by all waves.
__global__ void k_gemm_bias(const bf16_t* __restrict__ A, int K,
                            const bf16_t* __restrict__ W,
                            const float* __restrict__ biasF,
                            const float* __restrict__ biasB,
                            int N, bf16_t* __restrict__ out) {
  __shared__ bf16_t bt[2][128][32];  // 16 KB
  const int lane = threadIdx.x & 31;
  const int w = threadIdx.x >> 5;
  const int m0 = blockIdx.x * 128 + w * 16;
  const int bn = blockIdx.y * 128;
  const int KT = K >> 5;
  const bf16_t* Wb = W + (size_t)bn * K;
  if (w == 0)
    tdm_load_2d(Wb, lds_off_of(&bt[0][0][0]), 32u, 128u, (unsigned long long)K);
  v8f acc[8];
#pragma unroll
  for (int nt = 0; nt < 8; ++nt) acc[nt] = vzero8();
  for (int kt = 0; kt < KT; ++kt) {
    v16bf a = load_frag_a(A, K, m0, kt * 32, lane);   // overlaps with wait below
    if (w == 0) __builtin_amdgcn_s_wait_tensorcnt(0);
    __syncthreads();                                  // B tile [kt&1] visible
    if (w == 0 && kt + 1 < KT)
      tdm_load_2d(Wb + (kt + 1) * 32, lds_off_of(&bt[(kt + 1) & 1][0][0]),
                  32u, 128u, (unsigned long long)K);
    const bf16_t* bbase = &bt[kt & 1][0][0];
    v16bf bf[8];
#pragma unroll
    for (int nt = 0; nt < 8; ++nt) bf[nt] = load_frag_b(bbase, 32, nt * 16, 0, lane);
#pragma unroll
    for (int nt = 0; nt < 8; ++nt) acc[nt] = wmma_bf16(a, bf[nt], acc[nt]);
    __syncthreads();                                  // reads done before next DMA
  }
  const int lh = lane >> 4, ln = lane & 15;
#pragma unroll
  for (int nt = 0; nt < 8; ++nt) {
    int n = bn + nt * 16 + ln;
    float bias = (n < 768) ? biasF[n] : biasB[n - 768];
#pragma unroll
    for (int v = 0; v < 8; ++v) {
      int m = m0 + lh * 8 + v;
      out[(size_t)m * N + n] = f2bf(acc[nt][v] + bias);
    }
  }
}

// ---------- GRU scan ----------
// MT = M-tiles (seqs/16) per block; block = MT*128 threads; wave: mt=w>>2, q=w&3.
// h kept in LDS; per step GEMM h(16MT x 256) @ whh^T(256 x 768) on WMMA.
// xp gate slab for the NEXT step is TDM-prefetched into LDS during this step.
template<int MT>
__global__ void k_gru(const bf16_t* __restrict__ xp,   // [nSeq*L][1536]
                      const bf16_t* __restrict__ wh,   // [2][768][256] bf16
                      const float* __restrict__ bhhF,
                      const float* __restrict__ bhhB,
                      float* __restrict__ out,          // [nSeq*L][512]
                      bf16_t* __restrict__ outb,        // [nSeq*L][512]
                      int L) {
  __shared__ float  h32[MT * 16][256];
  __shared__ bf16_t hb[MT * 16][256];
  __shared__ bf16_t xbuf[2][MT * 16][768];
  const int tid = threadIdx.x;
  const int lane = tid & 31;
  const int w = tid >> 5;
  const int mt = w >> 2;
  const int q = w & 3;
  const int dir = blockIdx.y;
  const int seq0 = blockIdx.x * (MT * 16);
  const bf16_t* whd = wh + (size_t)dir * 768 * 256;
  const float* bhh = dir ? bhhB : bhhF;
  for (int i = tid; i < MT * 16 * 256; i += blockDim.x) {
    (&h32[0][0])[i] = 0.f;
    (&hb[0][0])[i] = f2bf(0.f);
  }
  __syncthreads();
  // prefetch xp slab for step 0
  if (w == 0) {
    int t0 = dir ? (L - 1) : 0;
    tdm_load_2d(xp + ((size_t)seq0 * L + t0) * 1536 + (size_t)dir * 768,
                lds_off_of(&xbuf[0][0][0]), 768u, (unsigned)(MT * 16),
                (unsigned long long)L * 1536ull);
  }
  const int lh = lane >> 4, ln = lane & 15;
  for (int step = 0; step < L; ++step) {
    const int t = dir ? (L - 1 - step) : step;
    v8f acc[3][4];
#pragma unroll
    for (int g = 0; g < 3; ++g)
#pragma unroll
      for (int nt = 0; nt < 4; ++nt) acc[g][nt] = vzero8();
#pragma unroll
    for (int kt = 0; kt < 8; ++kt) {
      v16bf a = load_frag_a(&hb[0][0], 256, mt * 16, kt * 32, lane);
#pragma unroll
      for (int g = 0; g < 3; ++g)
#pragma unroll
        for (int nt = 0; nt < 4; ++nt) {
          v16bf b = load_frag_b(whd, 256, g * 256 + q * 64 + nt * 16, kt * 32, lane);
          acc[g][nt] = wmma_bf16(a, b, acc[g][nt]);
        }
    }
    if (w == 0) __builtin_amdgcn_s_wait_tensorcnt(0);
    __syncthreads();  // GEMM reads of hb done; xp slab [step&1] visible
    if (w == 0 && step + 1 < L) {
      int tn = dir ? (L - 2 - step) : (step + 1);
      tdm_load_2d(xp + ((size_t)seq0 * L + tn) * 1536 + (size_t)dir * 768,
                  lds_off_of(&xbuf[(step + 1) & 1][0][0]), 768u,
                  (unsigned)(MT * 16), (unsigned long long)L * 1536ull);
    }
    const bf16_t (*xq)[768] = xbuf[step & 1];
#pragma unroll
    for (int nt = 0; nt < 4; ++nt) {
      const int j = q * 64 + nt * 16 + ln;
#pragma unroll
      for (int v = 0; v < 8; ++v) {
        const int m = mt * 16 + lh * 8 + v;
        const int seq = seq0 + m;
        float ir = bf2f(xq[m][j]);
        float iz = bf2f(xq[m][256 + j]);
        float ix = bf2f(xq[m][512 + j]);
        float hr = acc[0][nt][v] + bhh[j];
        float hz = acc[1][nt][v] + bhh[256 + j];
        float hn = acc[2][nt][v] + bhh[512 + j];
        float r = sigm(ir + hr);
        float z = sigm(iz + hz);
        float n = tanhf(ix + r * hn);
        float hNew = (1.f - z) * n + z * h32[m][j];
        h32[m][j] = hNew;
        hb[m][j] = f2bf(hNew);
        const size_t orow = ((size_t)seq * L + t) * 512 + (size_t)dir * 256 + j;
        out[orow] = hNew;
        outb[orow] = f2bf(hNew);
      }
    }
    __syncthreads();  // updates visible before next step's GEMM / DMA overwrite
  }
}

// ---------- attention scores: scores[m] += sum_n tanh(A@W^T + b)[m][n]*ctx[n] ----------
__global__ void k_attn_scores(const bf16_t* __restrict__ A,  // [M][512]
                              const bf16_t* __restrict__ W,  // [512][512]
                              const float* __restrict__ bias,
                              const float* __restrict__ ctx,
                              float* __restrict__ scores) {
  __shared__ bf16_t bt[2][128][32];
  const int lane = threadIdx.x & 31;
  const int w = threadIdx.x >> 5;
  const int m0 = blockIdx.x * 128 + w * 16;
  const int bn = blockIdx.y * 128;
  const bf16_t* Wb = W + (size_t)bn * 512;
  if (w == 0)
    tdm_load_2d(Wb, lds_off_of(&bt[0][0][0]), 32u, 128u, 512ull);
  v8f acc[8];
#pragma unroll
  for (int nt = 0; nt < 8; ++nt) acc[nt] = vzero8();
  for (int kt = 0; kt < 16; ++kt) {
    v16bf a = load_frag_a(A, 512, m0, kt * 32, lane);
    if (w == 0) __builtin_amdgcn_s_wait_tensorcnt(0);
    __syncthreads();
    if (w == 0 && kt + 1 < 16)
      tdm_load_2d(Wb + (kt + 1) * 32, lds_off_of(&bt[(kt + 1) & 1][0][0]),
                  32u, 128u, 512ull);
    const bf16_t* bbase = &bt[kt & 1][0][0];
    v16bf bf[8];
#pragma unroll
    for (int nt = 0; nt < 8; ++nt) bf[nt] = load_frag_b(bbase, 32, nt * 16, 0, lane);
#pragma unroll
    for (int nt = 0; nt < 8; ++nt) acc[nt] = wmma_bf16(a, bf[nt], acc[nt]);
    __syncthreads();
  }
  const int lh = lane >> 4, ln = lane & 15;
  float rs[8];
#pragma unroll
  for (int v = 0; v < 8; ++v) rs[v] = 0.f;
#pragma unroll
  for (int nt = 0; nt < 8; ++nt) {
    int n = bn + nt * 16 + ln;
    float bb = bias[n], cn = ctx[n];
#pragma unroll
    for (int v = 0; v < 8; ++v) rs[v] += tanhf(acc[nt][v] + bb) * cn;
  }
#pragma unroll
  for (int v = 0; v < 8; ++v) {
#pragma unroll
    for (int off = 1; off < 16; off <<= 1) rs[v] += __shfl_xor(rs[v], off, 32);
    if (ln == 0) atomicAdd(&scores[m0 + lh * 8 + v], rs[v]);
  }
}

// ---------- softmax over T + weighted sum ----------
__global__ void k_attn_combine(const float* __restrict__ O,  // [nSeg*L][512]
                               const float* __restrict__ scores,
                               float* __restrict__ outF,
                               bf16_t* __restrict__ outB,     // may be null
                               int L) {
  __shared__ float e[128];
  const int seg = blockIdx.x;
  const int tid = threadIdx.x;
  if (tid < L) e[tid] = scores[seg * L + tid];
  __syncthreads();
  float mx = -1e30f;
  for (int t = 0; t < L; ++t) mx = fmaxf(mx, e[t]);
  __syncthreads();
  if (tid < L) e[tid] = __expf(e[tid] - mx);
  __syncthreads();
  float s = 0.f;
  for (int t = 0; t < L; ++t) s += e[t];
  float inv = 1.f / s;
  for (int h = tid; h < 512; h += blockDim.x) {
    float a = 0.f;
    for (int t = 0; t < L; ++t) a += e[t] * O[((size_t)seg * L + t) * 512 + h];
    a *= inv;
    outF[(size_t)seg * 512 + h] = a;
    if (outB) outB[(size_t)seg * 512 + h] = f2bf(a);
  }
}

extern "C" void kernel_launch(void* const* d_in, const int* in_sizes, int n_in,
                              void* d_out, int out_size, void* d_ws, size_t ws_size,
                              hipStream_t stream) {
  (void)in_sizes; (void)n_in; (void)out_size; (void)ws_size;
  const float* tf       = (const float*)d_in[0];
  const float* w_wih_f  = (const float*)d_in[2];
  const float* w_whh_f  = (const float*)d_in[3];
  const float* w_bih_f  = (const float*)d_in[4];
  const float* w_bhh_f  = (const float*)d_in[5];
  const float* w_wih_b  = (const float*)d_in[6];
  const float* w_whh_b  = (const float*)d_in[7];
  const float* w_bih_b  = (const float*)d_in[8];
  const float* w_bhh_b  = (const float*)d_in[9];
  const float* s_wih_f  = (const float*)d_in[10];
  const float* s_whh_f  = (const float*)d_in[11];
  const float* s_bih_f  = (const float*)d_in[12];
  const float* s_bhh_f  = (const float*)d_in[13];
  const float* s_wih_b  = (const float*)d_in[14];
  const float* s_whh_b  = (const float*)d_in[15];
  const float* s_bih_b  = (const float*)d_in[16];
  const float* s_bhh_b  = (const float*)d_in[17];
  const float* w_attn_w = (const float*)d_in[18];
  const float* w_attn_b = (const float*)d_in[19];
  const float* w_ctx    = (const float*)d_in[20];
  const float* s_attn_w = (const float*)d_in[21];
  const float* s_attn_b = (const float*)d_in[22];
  const float* s_ctx    = (const float*)d_in[23];

  char* ws = (char*)d_ws;
  size_t off = 0;
  auto alloc = [&](size_t bytes) -> void* {
    void* p = ws + off;
    off = (off + bytes + 255) & ~(size_t)255;
    return p;
  };
  bf16_t* X     = (bf16_t*)alloc((size_t)32768 * 512 * 2);
  bf16_t* WI    = (bf16_t*)alloc((size_t)1536 * 512 * 2);
  bf16_t* WH    = (bf16_t*)alloc((size_t)2 * 768 * 256 * 2);
  bf16_t* ATTW  = (bf16_t*)alloc((size_t)512 * 512 * 2);
  bf16_t* WIS   = (bf16_t*)alloc((size_t)1536 * 512 * 2);
  bf16_t* WHS   = (bf16_t*)alloc((size_t)2 * 768 * 256 * 2);
  bf16_t* SATTW = (bf16_t*)alloc((size_t)512 * 512 * 2);
  bf16_t* XP    = (bf16_t*)alloc((size_t)32768 * 1536 * 2);
  float*  OW    = (float*) alloc((size_t)32768 * 512 * 4);
  bf16_t* OWB   = (bf16_t*)alloc((size_t)32768 * 512 * 2);
  float*  SC_W  = (float*) alloc((size_t)32768 * 4);
  float*  SV    = (float*) alloc((size_t)256 * 512 * 4);
  bf16_t* SVB   = (bf16_t*)alloc((size_t)256 * 512 * 2);
  bf16_t* XPS   = (bf16_t*)alloc((size_t)256 * 1536 * 2);
  float*  OS    = (float*) alloc((size_t)256 * 512 * 4);
  bf16_t* OSB   = (bf16_t*)alloc((size_t)256 * 512 * 2);
  float*  SC_S  = (float*) alloc((size_t)256 * 4);

  auto cvt = [&](const float* s, bf16_t* d, int n) {
    k_cvt<<<(n + 255) / 256, 256, 0, stream>>>(s, d, n);
  };

  // Stage 0: packing / conversion
  k_pack_x<<<65536, 256, 0, stream>>>(tf, X);
  cvt(w_wih_f, WI, 768 * 512);
  cvt(w_wih_b, WI + 768 * 512, 768 * 512);
  cvt(w_whh_f, WH, 768 * 256);
  cvt(w_whh_b, WH + 768 * 256, 768 * 256);
  cvt(w_attn_w, ATTW, 512 * 512);
  cvt(s_wih_f, WIS, 768 * 512);
  cvt(s_wih_b, WIS + 768 * 512, 768 * 512);
  cvt(s_whh_f, WHS, 768 * 256);
  cvt(s_whh_b, WHS + 768 * 256, 768 * 256);
  cvt(s_attn_w, SATTW, 512 * 512);
  k_zero<<<(32768 + 255) / 256, 256, 0, stream>>>(SC_W, 32768);
  k_zero<<<1, 256, 0, stream>>>(SC_S, 256);

  // Stage 1: word input projections (both dirs fused, bias added)
  k_gemm_bias<<<dim3(256, 12), 256, 0, stream>>>(X, 512, WI, w_bih_f, w_bih_b,
                                                 1536, XP);
  // Stage 2: word BiGRU scan (8 chunks x 32 seqs, 2 dirs)
  k_gru<2><<<dim3(8, 2), 256, 0, stream>>>(XP, WH, w_bhh_f, w_bhh_b, OW, OWB, 128);
  // Stage 3: word attention
  k_attn_scores<<<dim3(256, 4), 256, 0, stream>>>(OWB, ATTW, w_attn_b, w_ctx, SC_W);
  k_attn_combine<<<256, 256, 0, stream>>>(OW, SC_W, SV, SVB, 128);
  // Stage 4: sentence input projections
  k_gemm_bias<<<dim3(2, 12), 256, 0, stream>>>(SVB, 512, WIS, s_bih_f, s_bih_b,
                                               1536, XPS);
  // Stage 5: sentence BiGRU scan (16 seqs of length 16)
  k_gru<1><<<dim3(1, 2), 128, 0, stream>>>(XPS, WHS, s_bhh_f, s_bhh_b, OS, OSB, 16);
  // Stage 6: sentence attention -> d_out (16,512) fp32
  k_attn_scores<<<dim3(2, 4), 256, 0, stream>>>(OSB, SATTW, s_attn_b, s_ctx, SC_S);
  k_attn_combine<<<16, 256, 0, stream>>>(OS, SC_S, (float*)d_out, nullptr, 16);
}
